// AttentionCore_16028817949463
// MI455X (gfx1250) — compile-verified, matrix operand order checked by reading, not run on hardware
//
#include <hip/hip_runtime.h>

// Problem constants (match reference)
#define B_  2
#define T_  2048
#define D_  1024
#define H_  16
#define HD_ 64
#define M_  (B_*T_)

typedef __attribute__((ext_vector_type(16))) __bf16 bf16x16;
typedef __attribute__((ext_vector_type(8)))  float  f32x8;
typedef unsigned int u32x4 __attribute__((ext_vector_type(4)));
typedef int          i32x8 __attribute__((ext_vector_type(8)));
typedef int          i32x4 __attribute__((ext_vector_type(4)));

#if defined(__has_builtin)
#if __has_builtin(__builtin_amdgcn_tensor_load_to_lds) && __has_builtin(__builtin_amdgcn_s_wait_tensorcnt)
#define USE_TDM 1
#endif
#endif
#ifndef USE_TDM
#define USE_TDM 0
#endif

union ABRegs { bf16x16 v; unsigned int u[8]; };

__device__ __forceinline__ unsigned short f32_to_bf16_bits(float f) {
    unsigned int x = __float_as_uint(f);
    unsigned int r = 0x7FFFu + ((x >> 16) & 1u);   // round-to-nearest-even
    return (unsigned short)((x + r) >> 16);
}
__device__ __forceinline__ unsigned int pack_bf16x2(float lo, float hi) {
    return (unsigned int)f32_to_bf16_bits(lo) |
           ((unsigned int)f32_to_bf16_bits(hi) << 16);
}

// Load one 16x32 bf16 operand tile in the CDNA5 WMMA A/B VGPR layout.
// Per ISA 7.12.2: lane = row%16 (+16 for khalf=1), VGPR i holds K pair
// {base(i)+8*khalf, +1}, base(i) = 2i (i<4) else 16+2(i-4).
// Works for B when the matrix is stored column-major ([N,K] row-major).
// Also used on LDS pointers (addrspace inference -> ds_load).
__device__ __forceinline__ bf16x16 load_tile_pairs(const unsigned short* __restrict__ base,
                                                   int row, int stride, int k0, int khalf) {
    ABRegs r;
#pragma unroll
    for (int i = 0; i < 8; ++i) {
        int kk = ((i < 4) ? (2 * i) : (16 + 2 * (i - 4))) + khalf * 8;
        r.u[i] = *(const unsigned int*)(base + row * stride + k0 + kk);
    }
    return r.v;
}

__device__ __forceinline__ float rowmax16(float v) {
    v = fmaxf(v, __shfl_xor(v, 1));
    v = fmaxf(v, __shfl_xor(v, 2));
    v = fmaxf(v, __shfl_xor(v, 4));
    v = fmaxf(v, __shfl_xor(v, 8));
    return v;
}
__device__ __forceinline__ float rowsum16(float v) {
    v += __shfl_xor(v, 1);
    v += __shfl_xor(v, 2);
    v += __shfl_xor(v, 4);
    v += __shfl_xor(v, 8);
    return v;
}

#if USE_TDM
// Issue a 2-D TDM tile load (global -> LDS). Descriptor per CDNA5 D# layout:
// group0: count=1 | lds_addr[63:32] | global_addr[120:64] | type=2[127:126]
// group1: wg_mask=0, data_size=1 (2B), tensor_dim0/1, tile_dim0/1, dim0 stride.
// All fields are wave-uniform; force SGPR-ness with readfirstlane.
// This toolchain's builtin is the 6-arg form:
//   (uint32x4 g0, int32x8 g1, int32x4 g2, int32x4 g3, int32x8 pad, i32 cpol)
__device__ __forceinline__ void tdm_load_2d(unsigned ldsAddr, const void* gptr,
                                            unsigned tensor_d0, unsigned tensor_d1,
                                            unsigned tile_d0, unsigned tile_d1,
                                            unsigned stride0)
{
    unsigned long long ga = (unsigned long long)(uintptr_t)gptr;
    unsigned ga_lo = (unsigned)__builtin_amdgcn_readfirstlane((int)(ga & 0xFFFFFFFFu));
    unsigned ga_hi = (unsigned)__builtin_amdgcn_readfirstlane((int)(ga >> 32));
    unsigned lds   = (unsigned)__builtin_amdgcn_readfirstlane((int)ldsAddr);

    u32x4 g0;
    g0.x = 1u;                         // count=1 user descriptor
    g0.y = lds;                        // lds_addr
    g0.z = ga_lo;                      // global_addr[31:0]
    g0.w = ga_hi | 0x80000000u;        // global_addr[56:32] | type=2<<30

    i32x8 g1;
    g1[0] = (int)(1u << 16);                                     // data_size=1 (2 bytes)
    g1[1] = (int)((tensor_d0 & 0xFFFFu) << 16);                  // tensor_dim0[15:0]
    g1[2] = (int)((tensor_d0 >> 16) | ((tensor_d1 & 0xFFFFu) << 16));
    g1[3] = (int)((tensor_d1 >> 16) | ((tile_d0 & 0xFFFFu) << 16));
    g1[4] = (int)(tile_d1 & 0xFFFFu);                            // tile_dim1, tile_dim2=0
    g1[5] = (int)stride0;                                        // tensor_dim0_stride[31:0]
    g1[6] = 0;
    g1[7] = 0;

    i32x4 z4 = {0, 0, 0, 0};
    i32x8 z8 = {0, 0, 0, 0, 0, 0, 0, 0};
    __builtin_amdgcn_tensor_load_to_lds(g0, g1, z4, z4, z8, 0);
}
#endif

// ---------------- conversion kernels ----------------
__global__ void cvt_f32_bf16(const float* __restrict__ in, unsigned short* __restrict__ out, int n) {
    int i = blockIdx.x * blockDim.x + threadIdx.x;
    if (i < n) out[i] = f32_to_bf16_bits(in[i]);
}

// W [Kd,Nd] f32 row-major  ->  Wt [Nd,Kd] bf16 (== column-major of W)
__global__ void transpose_cvt(const float* __restrict__ W, unsigned short* __restrict__ Wt,
                              int Kd, int Nd) {
    int n = blockIdx.x * 16 + threadIdx.x;
    int k = blockIdx.y * 16 + threadIdx.y;
    if (n < Nd && k < Kd) Wt[n * Kd + k] = f32_to_bf16_bits(W[k * Nd + n]);
}

// ---------------- tiled GEMM via WMMA bf16 ----------------
// Each wave computes a 32x64 output block (2 M-tiles x 4 N-tiles): per
// 32-wide K-chunk: 2 A-tile + 4 B-tile loads feed 8 WMMAs (0.75 loads/WMMA).
// mode 0: store bf16 into Q/K head layout [B,H,T,HD]
// mode 1: store bf16 into V transposed layout [B,H,HD,T]
// mode 2: store f32 plain [M,N]
__global__ __launch_bounds__(128) void gemm32x64_wmma(
    const unsigned short* __restrict__ A,
    const unsigned short* __restrict__ Bt,
    const float* __restrict__ bias,
    void* __restrict__ Out,
    int Mdim, int Ndim, int Kd, int mode)
{
    int wave = (int)((blockIdx.x * blockDim.x + threadIdx.x) >> 5);
    int lane = threadIdx.x & 31;
    int lidx = lane & 15;
    int half = lane >> 4;
    int ntiles = Ndim >> 6;               // 64-wide N blocks
    int tm = wave / ntiles;               // 32-row M block
    int tn = wave % ntiles;
    if (tm >= (Mdim >> 5)) return;

    f32x8 acc[2][4];
#pragma unroll
    for (int mi = 0; mi < 2; ++mi)
#pragma unroll
        for (int j = 0; j < 4; ++j) acc[mi][j] = (f32x8){};

    int rowA0 = tm * 32 + lidx;
    int rowA1 = tm * 32 + 16 + lidx;

    for (int k0 = 0; k0 < Kd; k0 += 32) {
        // prefetch next K-chunk (global_prefetch_b8)
        if (k0 + 32 < Kd) {
            __builtin_prefetch(A + rowA0 * Kd + k0 + 32, 0, 3);
            __builtin_prefetch(A + rowA1 * Kd + k0 + 32, 0, 3);
#pragma unroll
            for (int j = 0; j < 4; ++j)
                __builtin_prefetch(Bt + (tn * 64 + j * 16 + lidx) * Kd + k0 + 32, 0, 3);
        }
        bf16x16 a0 = load_tile_pairs(A, rowA0, Kd, k0, half);
        bf16x16 a1 = load_tile_pairs(A, rowA1, Kd, k0, half);
#pragma unroll
        for (int j = 0; j < 4; ++j) {
            bf16x16 b = load_tile_pairs(Bt, tn * 64 + j * 16 + lidx, Kd, k0, half);
            acc[0][j] = __builtin_amdgcn_wmma_f32_16x16x32_bf16(false, a0, false, b,
                                                                (short)0, acc[0][j], false, false);
            acc[1][j] = __builtin_amdgcn_wmma_f32_16x16x32_bf16(false, a1, false, b,
                                                                (short)0, acc[1][j], false, false);
        }
    }

#pragma unroll
    for (int j = 0; j < 4; ++j) {
        int n = tn * 64 + j * 16 + lidx;
        float bcol = bias[n];
        int h = n / HD_, hd = n % HD_;
#pragma unroll
        for (int mi = 0; mi < 2; ++mi) {
#pragma unroll
            for (int r = 0; r < 8; ++r) {
                int m = tm * 32 + mi * 16 + r + 8 * half;  // C layout row ownership
                float v = acc[mi][j][r] + bcol;
                if (mode == 2) {
                    ((float*)Out)[m * Ndim + n] = v;
                } else {
                    int b_ = m / T_, t = m % T_;
                    unsigned short bb = f32_to_bf16_bits(v);
                    if (mode == 0)
                        ((unsigned short*)Out)[((b_ * H_ + h) * T_ + t) * HD_ + hd] = bb;
                    else
                        ((unsigned short*)Out)[((b_ * H_ + h) * HD_ + hd) * T_ + t] = bb;
                }
            }
        }
    }
}

// ---------------- flash attention core ----------------
// One wave = one 16-query tile of one (b,h). Online softmax over T keys in
// 32-key chunks. K/V blocks staged into per-wave LDS by the Tensor Data
// Mover, double-buffered (TDM for chunk c+1 in flight while consuming c).
__global__ __launch_bounds__(128) void attn_fa_wmma(
    const unsigned short* __restrict__ Q,   // [B,H,T,HD] bf16
    const unsigned short* __restrict__ Kh,  // [B,H,T,HD] bf16
    const unsigned short* __restrict__ Vt,  // [B,H,HD,T] bf16
    unsigned short* __restrict__ attnOut,   // [B*T, D] bf16
    float scale)
{
#if USE_TDM
    __shared__ unsigned short kTile[4][2][32 * HD_]; // per-wave double-buffered K block
    __shared__ unsigned short vTile[4][2][HD_ * 32]; // per-wave double-buffered V block
#endif
    __shared__ float sP[4][16][33];                  // per-wave P relayout buffer

    int wslot = threadIdx.x >> 5;
    int gw = blockIdx.x * 4 + wslot;
    int bh = gw / (T_ / 16);
    int qtile = gw % (T_ / 16);
    if (bh >= B_ * H_) return;
    int lane = threadIdx.x & 31;
    int lidx = lane & 15;
    int half = lane >> 4;
    int q0 = qtile * 16;

    const unsigned short* Qb = Q  + bh * T_ * HD_;
    const unsigned short* Kb = Kh + bh * T_ * HD_;
    const unsigned short* Vb = Vt + bh * HD_ * T_;

    // Q tile (16 x 64) resident in registers: two 16x32 A operands
    bf16x16 qa0 = load_tile_pairs(Qb, q0 + lidx, HD_, 0,  half);
    bf16x16 qa1 = load_tile_pairs(Qb, q0 + lidx, HD_, 32, half);

    f32x8 o[4];
#pragma unroll
    for (int nt = 0; nt < 4; ++nt) o[nt] = (f32x8){};
    float mrow[8], lrow[8];
#pragma unroll
    for (int r = 0; r < 8; ++r) { mrow[r] = -1e30f; lrow[r] = 0.0f; }

#if USE_TDM
    // Stage K (32 keys x 64 hd, row-contig hd) and V (64 hd x 32 keys,
    // row-contig keys) blocks for a chunk into LDS buffer `buf`.
    auto stage = [&](int key0, int buf) {
        tdm_load_2d((unsigned)(uintptr_t)&kTile[wslot][buf][0],
                    Kb + key0 * HD_,
                    /*tensor_d0=*/HD_, /*tensor_d1=*/T_,
                    /*tile_d0=*/HD_,   /*tile_d1=*/32,
                    /*stride0=*/HD_);
        tdm_load_2d((unsigned)(uintptr_t)&vTile[wslot][buf][0],
                    Vb + key0,
                    /*tensor_d0=*/T_,  /*tensor_d1=*/HD_,
                    /*tile_d0=*/32,    /*tile_d1=*/HD_,
                    /*stride0=*/T_);
    };
#endif

    // Consume one 32-key chunk whose K/V operands are read through `kbuf`/`vbuf`
    // with `kstr`/`vstr` row strides and K-offsets (global fallback).
    auto consume = [&](int key0,
                       const unsigned short* kbuf, int kstr, int krow0,
                       const unsigned short* vbuf, int vstr, int vkoff) {
        f32x8 s0 = {}, s1 = {};
        {
            bf16x16 kb0 = load_tile_pairs(kbuf, krow0 + lidx,      kstr, 0,  half);
            bf16x16 kb1 = load_tile_pairs(kbuf, krow0 + lidx,      kstr, 32, half);
            s0 = __builtin_amdgcn_wmma_f32_16x16x32_bf16(false, qa0, false, kb0, (short)0, s0, false, false);
            s0 = __builtin_amdgcn_wmma_f32_16x16x32_bf16(false, qa1, false, kb1, (short)0, s0, false, false);
            bf16x16 kb2 = load_tile_pairs(kbuf, krow0 + 16 + lidx, kstr, 0,  half);
            bf16x16 kb3 = load_tile_pairs(kbuf, krow0 + 16 + lidx, kstr, 32, half);
            s1 = __builtin_amdgcn_wmma_f32_16x16x32_bf16(false, qa0, false, kb2, (short)0, s1, false, false);
            s1 = __builtin_amdgcn_wmma_f32_16x16x32_bf16(false, qa1, false, kb3, (short)0, s1, false, false);
        }

#pragma unroll
        for (int r = 0; r < 8; ++r) {
            float a0 = s0[r] * scale;
            float a1 = s1[r] * scale;
            float mx = rowmax16(fmaxf(a0, a1));
            float mnew = fmaxf(mrow[r], mx);
            float alpha = __expf(mrow[r] - mnew);
            float p0 = __expf(a0 - mnew);
            float p1 = __expf(a1 - mnew);
            float rs = rowsum16(p0 + p1);
            lrow[r] = lrow[r] * alpha + rs;
            mrow[r] = mnew;
#pragma unroll
            for (int nt = 0; nt < 4; ++nt) o[nt][r] *= alpha;
            sP[wslot][r + 8 * half][lidx]      = p0;   // C-layout write
            sP[wslot][r + 8 * half][16 + lidx] = p1;
        }

        ABRegs pa;   // re-read P in A-operand layout
#pragma unroll
        for (int i = 0; i < 8; ++i) {
            int kk = ((i < 4) ? (2 * i) : (16 + 2 * (i - 4))) + half * 8;
            pa.u[i] = pack_bf16x2(sP[wslot][lidx][kk], sP[wslot][lidx][kk + 1]);
        }

#pragma unroll
        for (int nt = 0; nt < 4; ++nt) {
            bf16x16 vb = load_tile_pairs(vbuf, nt * 16 + lidx, vstr, vkoff, half);
            o[nt] = __builtin_amdgcn_wmma_f32_16x16x32_bf16(false, pa.v, false, vb,
                                                            (short)0, o[nt], false, false);
        }
        (void)key0;
    };

#if USE_TDM
    stage(0, 0);
    for (int key0 = 0; key0 < T_ - 32; key0 += 32) {
        int buf = (key0 >> 5) & 1;
        stage(key0 + 32, buf ^ 1);
        __builtin_amdgcn_s_wait_tensorcnt(2);   // oldest K/V pair complete
        __asm__ volatile("" ::: "memory");
        consume(key0, &kTile[wslot][buf][0], HD_, 0, &vTile[wslot][buf][0], 32, 0);
    }
    {
        int buf = ((T_ - 32) >> 5) & 1;
        __builtin_amdgcn_s_wait_tensorcnt(0);
        __asm__ volatile("" ::: "memory");
        consume(T_ - 32, &kTile[wslot][buf][0], HD_, 0, &vTile[wslot][buf][0], 32, 0);
    }
#else
    for (int key0 = 0; key0 < T_; key0 += 32)
        consume(key0, Kb, HD_, key0, Vb, T_, key0);
#endif

    // Normalize and emit attn in [B*T, D] bf16 for the output projection
    int b_ = bh / H_;
    int h  = bh % H_;
#pragma unroll
    for (int nt = 0; nt < 4; ++nt) {
#pragma unroll
        for (int r = 0; r < 8; ++r) {
            float v = o[nt][r] / (lrow[r] + 1e-20f);
            int token = q0 + r + 8 * half;
            int col = h * HD_ + nt * 16 + lidx;
            attnOut[(b_ * T_ + token) * D_ + col] = f32_to_bf16_bits(v);
        }
    }
}

// ---------------- host orchestration ----------------
extern "C" void kernel_launch(void* const* d_in, const int* in_sizes, int n_in,
                              void* d_out, int out_size, void* d_ws, size_t ws_size,
                              hipStream_t stream)
{
    const float* x  = (const float*)d_in[0];
    const float* Wq = (const float*)d_in[1];
    const float* bq = (const float*)d_in[2];
    const float* Wk = (const float*)d_in[3];
    const float* bk = (const float*)d_in[4];
    const float* Wv = (const float*)d_in[5];
    const float* bv = (const float*)d_in[6];
    const float* Wo = (const float*)d_in[7];
    const float* bo = (const float*)d_in[8];
    float* out = (float*)d_out;

    size_t off = 0;
    auto carve = [&](size_t bytes) -> void* {
        void* p = (char*)d_ws + off;
        off += (bytes + 255) & ~(size_t)255;
        return p;
    };
    unsigned short* xb    = (unsigned short*)carve((size_t)M_ * D_ * 2);
    unsigned short* Wqt   = (unsigned short*)carve((size_t)D_ * D_ * 2);
    unsigned short* Wkt   = (unsigned short*)carve((size_t)D_ * D_ * 2);
    unsigned short* Wvt   = (unsigned short*)carve((size_t)D_ * D_ * 2);
    unsigned short* Wot   = (unsigned short*)carve((size_t)D_ * D_ * 2);
    unsigned short* Qh    = (unsigned short*)carve((size_t)M_ * D_ * 2);
    unsigned short* Kx    = (unsigned short*)carve((size_t)M_ * D_ * 2);
    unsigned short* Vth   = (unsigned short*)carve((size_t)M_ * D_ * 2);
    unsigned short* attnb = (unsigned short*)carve((size_t)M_ * D_ * 2);

    // 1) x -> bf16
    {
        int n = M_ * D_;
        cvt_f32_bf16<<<(n + 255) / 256, 256, 0, stream>>>(x, xb, n);
    }
    // 2) weights -> transposed bf16
    {
        dim3 blk(16, 16), grd(D_ / 16, D_ / 16);
        transpose_cvt<<<grd, blk, 0, stream>>>(Wq, Wqt, D_, D_);
        transpose_cvt<<<grd, blk, 0, stream>>>(Wk, Wkt, D_, D_);
        transpose_cvt<<<grd, blk, 0, stream>>>(Wv, Wvt, D_, D_);
        transpose_cvt<<<grd, blk, 0, stream>>>(Wo, Wot, D_, D_);
    }
    // 3) Q/K/V projections: (M/32)*(D/64) = 2048 waves -> 512 blocks
    {
        int blocks = (M_ / 32) * (D_ / 64) / 4;
        gemm32x64_wmma<<<blocks, 128, 0, stream>>>(xb, Wqt, bq, Qh,  M_, D_, D_, 0);
        gemm32x64_wmma<<<blocks, 128, 0, stream>>>(xb, Wkt, bk, Kx,  M_, D_, D_, 0);
        gemm32x64_wmma<<<blocks, 128, 0, stream>>>(xb, Wvt, bv, Vth, M_, D_, D_, 1);
    }
    // 4) flash attention core: B*H*(T/16) = 4096 waves -> 1024 blocks
    {
        int blocks = (B_ * H_ * (T_ / 16)) / 4;
        attn_fa_wmma<<<blocks, 128, 0, stream>>>(Qh, Kx, Vth, attnb, 0.125f /* 1/sqrt(64) */);
    }
    // 5) output projection (f32 result)
    {
        int blocks = (M_ / 32) * (D_ / 64) / 4;
        gemm32x64_wmma<<<blocks, 128, 0, stream>>>(attnb, Wot, bo, (void*)out, M_, D_, D_, 2);
    }
}
